// ComplexCrossAttention_8169027797674
// MI455X (gfx1250) — compile-verified, matrix operand order checked by reading, not run on hardware
//
#include <hip/hip_runtime.h>

// ---------------------------------------------------------------------------
// ComplexCrossAttention on MI455X (gfx1250, wave32, WMMA bf16 16x16x32)
// ---------------------------------------------------------------------------
//   1) pack inputs  -> X2 = [xr | xi]                  (bf16, rows x 2C)
//   2) pack weights -> W2t = transpose([[wr,wi],[-wi,wr]]) (bf16, 2C x 2C)
//      (transposed so GEMM B-tiles are contiguous-row copies -> async DMA)
//   3) Q2/K2/V2 projections: WMMA GEMM with global_load_async_to_lds_b128
//      ping-pong double buffering (ASYNCcnt-tracked, s_wait_asynccnt)
//   4) flash attention per (b,h): WMMA scores + online softmax + WMMA PV
//   5) Y2 = O2 @ W2o + bias, split-written as f32 [2,B,L,C]
// ---------------------------------------------------------------------------

typedef unsigned short u16;
typedef __attribute__((ext_vector_type(16))) __bf16 v16bf;
typedef __attribute__((ext_vector_type(8)))  float  v8f;

constexpr int   kB  = 2;
constexpr int   kL  = 2048;
constexpr int   kS  = 2048;
constexpr int   kC  = 1024;
constexpr int   kC2 = 2048;   // 2*C (real|imag concatenated)
constexpr int   kH  = 16;
constexpr int   kD  = 64;
constexpr float kScale = 0.125f;   // 1/sqrt(64)

// ---- helpers ---------------------------------------------------------------

static __device__ inline u16 f32_to_bf16(float f) {
  unsigned int u = __float_as_uint(f);
  u += 0x7FFFu + ((u >> 16) & 1u);          // round to nearest even
  return (u16)(u >> 16);
}

static __device__ inline unsigned pack2bf(float a, float b) {
  return (unsigned)f32_to_bf16(a) | ((unsigned)f32_to_bf16(b) << 16);
}

// Async DMA of 16 contiguous bytes global -> LDS (per lane), ASYNCcnt-tracked.
// VDST holds the LDS byte address (low 32 bits of the generic pointer).
static __device__ inline void async_copy_b128(const u16* gptr, u16* lptr) {
  unsigned lds_addr = (unsigned)(unsigned long long)(uintptr_t)lptr;
  asm volatile("global_load_async_to_lds_b128 %0, %1, off"
               :: "v"(lds_addr), "v"(gptr)
               : "memory");
}

// Load a 16x16x32 bf16 WMMA operand fragment: per ISA 7.12.2 the per-lane
// 16 elements are two runs of 8 contiguous K values (K = g*16 + half*8 + j%8).
static __device__ inline v16bf load_frag(const u16* p0, const u16* p1) {
  union U { uint4 q; __bf16 h[8]; } t0, t1;
  t0.q = *reinterpret_cast<const uint4*>(p0);
  t1.q = *reinterpret_cast<const uint4*>(p1);
  v16bf a;
#pragma unroll
  for (int j = 0; j < 8; ++j) { a[j] = t0.h[j]; a[8 + j] = t1.h[j]; }
  return a;
}

static __device__ inline v8f wmma_bf16(v16bf a, v16bf b, v8f c) {
  return __builtin_amdgcn_wmma_f32_16x16x32_bf16(
      false, a, false, b, (short)0, c, false, false);
}

// ---- 1) pack activations: X2[row][c]=xr, X2[row][C+c]=xi (bf16) ------------

__global__ void cca_pack_x(const float* __restrict__ xr,
                           const float* __restrict__ xi,
                           u16* __restrict__ X2, int rows) {
  int idx = blockIdx.x * blockDim.x + threadIdx.x;
  int total = rows * kC;
  if (idx >= total) return;
  int row = idx / kC, c = idx - row * kC;
  X2[(size_t)row * kC2 + c]      = f32_to_bf16(xr[idx]);
  X2[(size_t)row * kC2 + kC + c] = f32_to_bf16(xi[idx]);
}

// ---- 2) pack weights TRANSPOSED: W2t[n][k] = W2[k][n] ----------------------
// W2 = [[wr, wi], [-wi, wr]]  =>
//   W2t[n   ][k   ] =  wr[k][n]     W2t[n   ][C+k ] = -wi[k][n]
//   W2t[C+n ][k   ] =  wi[k][n]     W2t[C+n ][C+k ] =  wr[k][n]
// One thread handles a k-pair so each store is a packed u32.

__global__ void cca_pack_wT(const float* __restrict__ wr,
                            const float* __restrict__ wi,
                            u16* __restrict__ W2t) {
  int idx = blockIdx.x * blockDim.x + threadIdx.x;
  if (idx >= (kC / 2) * kC) return;
  int kp = idx / kC;                 // k-pair index
  int n  = idx - kp * kC;
  int k  = kp * 2;
  float r0 = wr[(size_t)k * kC + n], r1 = wr[(size_t)(k + 1) * kC + n];
  float i0 = wi[(size_t)k * kC + n], i1 = wi[(size_t)(k + 1) * kC + n];
  unsigned pr = pack2bf(r0, r1);
  unsigned pi = pack2bf(i0, i1);
  unsigned pni = pack2bf(-i0, -i1);
  *reinterpret_cast<unsigned*>(&W2t[(size_t)n * kC2 + k])             = pr;
  *reinterpret_cast<unsigned*>(&W2t[(size_t)n * kC2 + kC + k])        = pni;
  *reinterpret_cast<unsigned*>(&W2t[(size_t)(kC + n) * kC2 + k])      = pi;
  *reinterpret_cast<unsigned*>(&W2t[(size_t)(kC + n) * kC2 + kC + k]) = pr;
}

// ---- 3/5) bf16 WMMA GEMM with async-to-LDS double buffering ----------------
// out[rows x 2C] = A[rows x 2C] @ W2t^T  (W2t stored n-major)
// mode 0: bf16 out (rows x 2C); mode 1: split f32 [2,B,L,C] + bias.
// 128x128x32 tiles, 256 threads = 8 waves (4 along M x 2 along N),
// each wave computes a 32x64 slab = 2x4 WMMA accumulators.

#define GBM 128
#define GBN 128
#define GBK 32

__global__ __launch_bounds__(256)
void cca_gemm_bf16(const u16* __restrict__ A, const u16* __restrict__ Wt,
                   u16* __restrict__ outBf, float* __restrict__ outF,
                   const float* __restrict__ biasR,
                   const float* __restrict__ biasI,
                   int rows, int mode) {
  __shared__ __align__(16) u16 As[2][GBM][GBK];   // (m,k), 8 KB x2
  __shared__ __align__(16) u16 Bs[2][GBN][GBK];   // (n,k), 8 KB x2

  const int tid  = threadIdx.x;
  const int lane = tid & 31, wave = tid >> 5;
  const int half = lane >> 4, l16 = lane & 15;
  const int wm = wave & 3;        // 4 waves along M (32 rows each)
  const int wn = wave >> 2;       // 2 waves along N (64 cols each)
  const int rowBase = blockIdx.x * GBM;
  const int colBase = blockIdx.y * GBN;

  // issue this thread's 4 async b128 copies (2 for A, 2 for B) for one tile
  auto issue_tile = [&](int buf, int k0) {
#pragma unroll
    for (int i = 0; i < 2; ++i) {
      int c = tid + i * 256;               // 0..511 chunks of 8 u16
      int r = c >> 2, off = (c & 3) * 8;
      async_copy_b128(&A[(size_t)(rowBase + r) * kC2 + k0 + off],
                      &As[buf][r][off]);
    }
#pragma unroll
    for (int i = 0; i < 2; ++i) {
      int c = tid + i * 256;
      int r = c >> 2, off = (c & 3) * 8;
      async_copy_b128(&Wt[(size_t)(colBase + r) * kC2 + k0 + off],
                      &Bs[buf][r][off]);
    }
  };

  v8f acc[2][4];
#pragma unroll
  for (int mi = 0; mi < 2; ++mi)
#pragma unroll
    for (int ni = 0; ni < 4; ++ni) acc[mi][ni] = (v8f)(0.0f);

  issue_tile(0, 0);
  issue_tile(1, GBK);

  const int nK = kC2 / GBK;   // 64 K-steps
  for (int i = 0; i < nK; ++i) {
    const int cur = i & 1;
    // current tile's 4 copies landed; next tile's 4 may stay in flight
    if (i < nK - 1) asm volatile("s_wait_asynccnt 0x4" ::: "memory");
    else            asm volatile("s_wait_asynccnt 0x0" ::: "memory");
    __syncthreads();

    v16bf af[2], bfr[4];
#pragma unroll
    for (int mi = 0; mi < 2; ++mi) {
      int m = wm * 32 + mi * 16 + l16;
      af[mi] = load_frag(&As[cur][m][half * 8], &As[cur][m][16 + half * 8]);
    }
#pragma unroll
    for (int ni = 0; ni < 4; ++ni) {
      int n = wn * 64 + ni * 16 + l16;
      bfr[ni] = load_frag(&Bs[cur][n][half * 8], &Bs[cur][n][16 + half * 8]);
    }
#pragma unroll
    for (int mi = 0; mi < 2; ++mi)
#pragma unroll
      for (int ni = 0; ni < 4; ++ni)
        acc[mi][ni] = wmma_bf16(af[mi], bfr[ni], acc[mi][ni]);

    __syncthreads();                      // all waves done reading buf `cur`
    if (i + 2 < nK) issue_tile(cur, (i + 2) * GBK);
  }

  // epilogue (C/D layout: vgpr r -> row r (lanes 0-15) / r+8 (lanes 16-31))
#pragma unroll
  for (int mi = 0; mi < 2; ++mi)
#pragma unroll
    for (int ni = 0; ni < 4; ++ni)
#pragma unroll
      for (int r = 0; r < 8; ++r) {
        int grow = rowBase + wm * 32 + mi * 16 + r + half * 8;
        int gcol = colBase + wn * 64 + ni * 16 + l16;
        float v = acc[mi][ni][r];
        if (mode == 0) {
          outBf[(size_t)grow * kC2 + gcol] = f32_to_bf16(v);
        } else {
          if (gcol < kC)
            outF[(size_t)grow * kC + gcol] = v + biasR[gcol];
          else
            outF[(size_t)rows * kC + (size_t)grow * kC + (gcol - kC)] =
                v + biasI[gcol - kC];
        }
      }
}

// ---- 4) flash attention, one (b,h) x 64 query rows per block ---------------
// 128 threads = 4 waves; wave w owns query rows [l0+16w, l0+16w+16).
// Q fragments live in registers for the whole S loop.  K is consumed as the
// WMMA B operand straight from global (head dim contiguous).  V is staged
// transposed in LDS with register batching so next-chunk global loads overlap
// current-chunk WMMAs.  P flips C-layout -> A-layout through LDS.

__global__ __launch_bounds__(128)
void cca_attention(const u16* __restrict__ Q2, const u16* __restrict__ K2,
                   const u16* __restrict__ V2, u16* __restrict__ O2) {
  __shared__ __align__(16) u16 VtR[kD][64];     // (d, s) 8 KB
  __shared__ __align__(16) u16 VtI[kD][64];     // 8 KB
  __shared__ __align__(16) u16 Ps[4][16][64];   // per-wave P tiles, 8 KB

  const int tid  = threadIdx.x;
  const int lane = tid & 31, wave = tid >> 5;
  const int half = lane >> 4, l16 = lane & 15;
  const int b  = blockIdx.x >> 4;
  const int h  = blockIdx.x & 15;
  const int l0 = blockIdx.y * 64;
  const int qrow0 = b * kL + l0 + wave * 16;    // this wave's 16 query rows
  const int dcol  = h * kD;

  // Q fragments (qr & qi), 2 K-steps of 32 over the 64-wide head dim
  v16bf aqr[2], aqi[2];
#pragma unroll
  for (int ks = 0; ks < 2; ++ks) {
    const u16* qp = Q2 + (size_t)(qrow0 + l16) * kC2 + dcol + ks * 32 + half * 8;
    aqr[ks] = load_frag(qp, qp + 16);
    const u16* qpi = qp + kC;
    aqi[ks] = load_frag(qpi, qpi + 16);
  }

  // register-staged V chunk (batched global loads)
  unsigned vreg_r[16], vreg_i[16];
  auto loadV = [&](int s0) {
#pragma unroll
    for (int i = 0; i < 16; ++i) {
      int p  = tid + i * 128;                  // 0..2047 d-pairs
      int s  = p >> 5;
      int dd = (p & 31) * 2;
      size_t g = (size_t)(b * kS + s0 + s) * kC2 + dcol + dd;
      vreg_r[i] = *reinterpret_cast<const unsigned*>(&V2[g]);
      vreg_i[i] = *reinterpret_cast<const unsigned*>(&V2[g + kC]);
    }
  };
  auto storeV = [&]() {
#pragma unroll
    for (int i = 0; i < 16; ++i) {
      int p  = tid + i * 128;
      int s  = p >> 5;
      int dd = (p & 31) * 2;
      VtR[dd][s] = (u16)(vreg_r[i] & 0xFFFFu);
      VtR[dd + 1][s] = (u16)(vreg_r[i] >> 16);
      VtI[dd][s] = (u16)(vreg_i[i] & 0xFFFFu);
      VtI[dd + 1][s] = (u16)(vreg_i[i] >> 16);
    }
  };

  float mrun[8], lrun[8];
  v8f oR[4], oI[4];
#pragma unroll
  for (int r = 0; r < 8; ++r) { mrun[r] = -1e30f; lrun[r] = 0.0f; }
#pragma unroll
  for (int nt = 0; nt < 4; ++nt) { oR[nt] = (v8f)(0.0f); oI[nt] = (v8f)(0.0f); }

  loadV(0);
  for (int s0 = 0; s0 < kS; s0 += 64) {
    __syncthreads();                 // previous PV done reading Vt
    storeV();
    __syncthreads();
    if (s0 + 64 < kS) loadV(s0 + 64);   // overlap with this chunk's WMMAs

    // scores: 4 n-tiles x (2 ksteps x {qr.kr, qi.ki}) = 16 WMMAs
    v8f sc[4];
#pragma unroll
    for (int nt = 0; nt < 4; ++nt) sc[nt] = (v8f)(0.0f);
#pragma unroll
    for (int nt = 0; nt < 4; ++nt) {
      size_t srow = (size_t)(b * kS + s0 + nt * 16 + l16) * kC2 + dcol;
#pragma unroll
      for (int ks = 0; ks < 2; ++ks) {
        const u16* kp = K2 + srow + ks * 32 + half * 8;
        sc[nt] = wmma_bf16(aqr[ks], load_frag(kp, kp + 16), sc[nt]);
        const u16* kpi = kp + kC;
        sc[nt] = wmma_bf16(aqi[ks], load_frag(kpi, kpi + 16), sc[nt]);
      }
    }

    // online softmax (rows live in 16-lane halves; xor-reduce within half)
#pragma unroll
    for (int r = 0; r < 8; ++r) {
      float v0 = sc[0][r] * kScale, v1 = sc[1][r] * kScale;
      float v2 = sc[2][r] * kScale, v3 = sc[3][r] * kScale;
      float rm = fmaxf(fmaxf(v0, v1), fmaxf(v2, v3));
#pragma unroll
      for (int off = 8; off >= 1; off >>= 1) rm = fmaxf(rm, __shfl_xor(rm, off, 32));
      float mnew = fmaxf(mrun[r], rm);
      float corr = __expf(mrun[r] - mnew);
      float p0 = __expf(v0 - mnew), p1 = __expf(v1 - mnew);
      float p2 = __expf(v2 - mnew), p3 = __expf(v3 - mnew);
      float rs = (p0 + p1) + (p2 + p3);
#pragma unroll
      for (int off = 8; off >= 1; off >>= 1) rs += __shfl_xor(rs, off, 32);
      lrun[r] = lrun[r] * corr + rs;
      mrun[r] = mnew;
#pragma unroll
      for (int nt = 0; nt < 4; ++nt) { oR[nt][r] *= corr; oI[nt][r] *= corr; }
      int mrow = r + half * 8;
      Ps[wave][mrow][0 * 16 + l16] = f32_to_bf16(p0);
      Ps[wave][mrow][1 * 16 + l16] = f32_to_bf16(p1);
      Ps[wave][mrow][2 * 16 + l16] = f32_to_bf16(p2);
      Ps[wave][mrow][3 * 16 + l16] = f32_to_bf16(p3);
    }
    __syncthreads();

    // O += P @ V  (2 ksteps x 4 n-tiles x {r,i} = 16 WMMAs)
#pragma unroll
    for (int ks = 0; ks < 2; ++ks) {
      const u16* pp = &Ps[wave][l16][ks * 32 + half * 8];
      v16bf ap = load_frag(pp, pp + 16);
#pragma unroll
      for (int nt = 0; nt < 4; ++nt) {
        const u16* vp = &VtR[nt * 16 + l16][ks * 32 + half * 8];
        oR[nt] = wmma_bf16(ap, load_frag(vp, vp + 16), oR[nt]);
        const u16* vpi = &VtI[nt * 16 + l16][ks * 32 + half * 8];
        oI[nt] = wmma_bf16(ap, load_frag(vpi, vpi + 16), oI[nt]);
      }
    }
  }

  // normalize and emit O2 = [out_r | out_i] (bf16, rows x 2C)
#pragma unroll
  for (int r = 0; r < 8; ++r) {
    float inv = 1.0f / lrun[r];
    size_t grow = (size_t)(qrow0 + r + half * 8) * kC2 + dcol;
#pragma unroll
    for (int nt = 0; nt < 4; ++nt) {
      O2[grow + nt * 16 + l16]      = f32_to_bf16(oR[nt][r] * inv);
      O2[grow + kC + nt * 16 + l16] = f32_to_bf16(oI[nt][r] * inv);
    }
  }
}

// ---------------------------------------------------------------------------

extern "C" void kernel_launch(void* const* d_in, const int* in_sizes, int n_in,
                              void* d_out, int out_size, void* d_ws, size_t ws_size,
                              hipStream_t stream) {
  const float* in_r  = (const float*)d_in[0];
  const float* in_i  = (const float*)d_in[1];
  const float* ctx_r = (const float*)d_in[2];
  const float* ctx_i = (const float*)d_in[3];
  const float* wq_r  = (const float*)d_in[4];
  const float* wq_i  = (const float*)d_in[5];
  const float* wk_r  = (const float*)d_in[6];
  const float* wk_i  = (const float*)d_in[7];
  const float* wv_r  = (const float*)d_in[8];
  const float* wv_i  = (const float*)d_in[9];
  const float* wo_r  = (const float*)d_in[10];
  const float* wo_i  = (const float*)d_in[11];
  const float* bo_r  = (const float*)d_in[12];
  const float* bo_i  = (const float*)d_in[13];
  float* out = (float*)d_out;

  const int rowsL = kB * kL;                 // 4096
  const int rowsS = kB * kS;                 // 4096
  const size_t E_X = (size_t)rowsL * kC2;    // activation buffer elems (bf16)
  const size_t E_W = (size_t)kC2 * kC2;      // packed weight elems (bf16)

  u16* p     = (u16*)d_ws;
  u16* X2in  = p; p += E_X;
  u16* X2ctx = p; p += E_X;
  u16* Wq2   = p; p += E_W;
  u16* Wk2   = p; p += E_W;
  u16* Wv2   = p; p += E_W;
  u16* Wo2   = p; p += E_W;
  u16* Q2    = p; p += E_X;
  u16* K2    = p; p += E_X;
  u16* V2    = p; p += E_X;
  u16* O2    = p;

  // 1) pack activations
  {
    int total = rowsL * kC;
    cca_pack_x<<<(total + 255) / 256, 256, 0, stream>>>(in_r,  in_i,  X2in,  rowsL);
    cca_pack_x<<<(total + 255) / 256, 256, 0, stream>>>(ctx_r, ctx_i, X2ctx, rowsS);
  }
  // 2) pack weights (transposed)
  {
    int total = (kC / 2) * kC;
    int blk = (total + 255) / 256;
    cca_pack_wT<<<blk, 256, 0, stream>>>(wq_r, wq_i, Wq2);
    cca_pack_wT<<<blk, 256, 0, stream>>>(wk_r, wk_i, Wk2);
    cca_pack_wT<<<blk, 256, 0, stream>>>(wv_r, wv_i, Wv2);
    cca_pack_wT<<<blk, 256, 0, stream>>>(wo_r, wo_i, Wo2);
  }
  // 3) projections (bf16 WMMA GEMMs, async double-buffered)
  {
    dim3 grid(rowsL / GBM, kC2 / GBN);
    cca_gemm_bf16<<<grid, 256, 0, stream>>>(X2in,  Wq2, Q2, nullptr, nullptr, nullptr, rowsL, 0);
    cca_gemm_bf16<<<grid, 256, 0, stream>>>(X2ctx, Wk2, K2, nullptr, nullptr, nullptr, rowsS, 0);
    cca_gemm_bf16<<<grid, 256, 0, stream>>>(X2ctx, Wv2, V2, nullptr, nullptr, nullptr, rowsS, 0);
  }
  // 4) attention
  {
    dim3 grid(kB * kH, kL / 64);
    cca_attention<<<grid, 128, 0, stream>>>(Q2, K2, V2, O2);
  }
  // 5) output projection + bias, split-write f32 [2,B,L,C]
  {
    dim3 grid(rowsL / GBM, kC2 / GBN);
    cca_gemm_bf16<<<grid, 256, 0, stream>>>(O2, Wo2, nullptr, out, bo_r, bo_i, rowsL, 1);
  }
}
